// s4ormer_lm_63247688401502
// MI455X (gfx1250) — compile-verified, matrix operand order checked by reading, not run on hardware
//
#include <hip/hip_runtime.h>
#include <hip/hip_bf16.h>

// ---------------------------------------------------------------------------
// S4ormer LM forward for MI455X (gfx1250, wave32, WMMA).
// - All heavy GEMMs: bf16 v_wmma_f32_16x16x32_bf16, double-buffered LDS tiles
//   filled via CDNA5 async DMA (global_load_async_to_lds_b128 / ASYNCcnt).
// - S4 layer: equivalent diagonal-SSM recurrence, one wave per channel,
//   one complex mode per lane (NSTATE == wave32 width).
// ---------------------------------------------------------------------------

#define DIMc   1024
#define DEPTHc 4
#define NS4c   8
#define NSTc   32
#define VOCc   32000
#define Bc     4
#define Lc     1024
#define FFc    4096
#define Mrows  (Bc * Lc)   // 4096 token positions

typedef __attribute__((ext_vector_type(16))) __bf16 bf16x16;
typedef __attribute__((ext_vector_type(8)))  float  f32x8;

__device__ __forceinline__ unsigned short f2bf(float f) {
  unsigned int u = __float_as_uint(f);
  u += 0x7FFFu + ((u >> 16) & 1u);          // round to nearest even
  return (unsigned short)(u >> 16);
}

// CDNA5 async copy: 16B global -> LDS, tracked with ASYNCcnt (no VGPR bounce).
__device__ __forceinline__ void async_cp16(const void* g, void* l) {
  unsigned       lo = (unsigned)(uintptr_t)l;              // LDS byte offset
  unsigned long long ga = (unsigned long long)(uintptr_t)g;
  asm volatile("global_load_async_to_lds_b128 %0, %1, off"
               :: "v"(lo), "v"(ga) : "memory");
}
__device__ __forceinline__ void wait_async0() {
  asm volatile("s_wait_asynccnt 0x0" ::: "memory");
}

// ------------------------- weight fp32 -> bf16 -----------------------------
__global__ void cast_f32_bf16_k(const float* __restrict__ in,
                                unsigned short* __restrict__ out, size_t n) {
  size_t i = (size_t)blockIdx.x * blockDim.x + threadIdx.x;
  size_t stride = (size_t)gridDim.x * blockDim.x;
  for (; i < n; i += stride) out[i] = f2bf(in[i]);
}

// ------------------------- embedding gather --------------------------------
__global__ void gather_k(const int* __restrict__ x, const float* __restrict__ emb,
                         float* __restrict__ h) {
  int bl = blockIdx.x;
  int tok = x[bl];
  const float* e = emb + (size_t)tok * DIMc;
  float* o = h + (size_t)bl * DIMc;
  for (int d = threadIdx.x; d < DIMc; d += blockDim.x) o[d] = e[d];
}

// ------------------------- residual add ------------------------------------
__global__ void add_k(float* __restrict__ h, const float* __restrict__ r, size_t n) {
  size_t i = (size_t)blockIdx.x * blockDim.x + threadIdx.x;
  size_t stride = (size_t)gridDim.x * blockDim.x;
  for (; i < n; i += stride) h[i] += r[i];
}

// ------------------------- LayerNorm (row = token) -------------------------
__global__ __launch_bounds__(256)
void layernorm_k(const float* __restrict__ x, const float* __restrict__ w,
                 const float* __restrict__ b, float* __restrict__ yf,
                 unsigned short* __restrict__ yb) {
  const int row = blockIdx.x;
  const float* xr = x + (size_t)row * DIMc;
  __shared__ float s1[256], s2[256];
  float sum = 0.f, sq = 0.f;
  for (int d = threadIdx.x; d < DIMc; d += 256) {
    float v = xr[d]; sum += v; sq += v * v;
  }
  s1[threadIdx.x] = sum; s2[threadIdx.x] = sq;
  __syncthreads();
  for (int off = 128; off > 0; off >>= 1) {
    if (threadIdx.x < off) {
      s1[threadIdx.x] += s1[threadIdx.x + off];
      s2[threadIdx.x] += s2[threadIdx.x + off];
    }
    __syncthreads();
  }
  float mean = s1[0] * (1.f / DIMc);
  float var  = s2[0] * (1.f / DIMc) - mean * mean;
  float rstd = rsqrtf(var + 1e-5f);
  for (int d = threadIdx.x; d < DIMc; d += 256) {
    float v = (xr[d] - mean) * rstd * w[d] + b[d];
    if (yf) yf[(size_t)row * DIMc + d] = v;
    if (yb) yb[(size_t)row * DIMc + d] = f2bf(v);
  }
}

// ------------------------- S4D recurrence + GELU ---------------------------
// One wave per (batch, channel); lane n owns complex mode n (NSTATE == 32).
// s[t] = lambda*s[t-1] + u[t];  y[t] = 2*Re(sum_n dC_n s_n[t]) + D*u[t]
// Equivalent to the reference's ZOH-discretized FFT convolution.
__global__ __launch_bounds__(256)
void s4_scan_k(const float* __restrict__ u, const float* __restrict__ log_dt,
               const float* __restrict__ A_re, const float* __restrict__ A_im,
               const float* __restrict__ C_re, const float* __restrict__ C_im,
               const float* __restrict__ Dp, unsigned short* __restrict__ ybf) {
  const int wid  = (int)((blockIdx.x * blockDim.x + threadIdx.x) >> 5);
  const int lane = threadIdx.x & 31;
  if (wid >= Bc * DIMc) return;
  const int b = wid / DIMc;
  const int h = wid % DIMc;

  const float dt = __expf(log_dt[h]);
  const float ar = A_re[h * NSTc + lane];
  const float ai = A_im[h * NSTc + lane];
  const float dtar = dt * ar, dtai = dt * ai;
  const float em = __expf(dtar);
  const float er = em * __cosf(dtai);       // lambda = exp(dt*A)
  const float ei = em * __sinf(dtai);
  // dC = C * (lambda - 1) / A
  const float nr = er - 1.f, ni = ei;
  const float den = 1.f / (ar * ar + ai * ai);
  const float fr = (nr * ar + ni * ai) * den;
  const float fi = (ni * ar - nr * ai) * den;
  const float cr = C_re[h * NSTc + lane];
  const float ci = C_im[h * NSTc + lane];
  const float dcr = cr * fr - ci * fi;
  const float dci = cr * fi + ci * fr;
  const float Dv = Dp[h];

  float sr = 0.f, si = 0.f;
  const float* ub = u + (size_t)b * Lc * DIMc + h;
  unsigned short* yb = ybf + (size_t)b * Lc * DIMc + h;

  for (int t = 0; t < Lc; ++t) {
    if (t + 16 < Lc) __builtin_prefetch(&ub[(size_t)(t + 16) * DIMc], 0, 1);
    float ut = ub[(size_t)t * DIMc];
    float nsr = er * sr - ei * si + ut;
    float nsi = er * si + ei * sr;
    sr = nsr; si = nsi;
    float c = dcr * sr - dci * si;          // Re(dC * s)
    #pragma unroll
    for (int off = 16; off > 0; off >>= 1) c += __shfl_xor(c, off, 32);
    float yv = 2.f * c + Dv * ut;
    // tanh-approx GELU (jax.nn.gelu default)
    float g = 0.5f * yv * (1.f + tanhf(0.7978845608028654f *
                                       (yv + 0.044715f * yv * yv * yv)));
    if (lane == 0) yb[(size_t)t * DIMc] = f2bf(g);
  }
}

// ------------------------- bf16 WMMA GEMM ----------------------------------
// C[m,n] = sum_k A[m,k] * W[n,k]  (TN: both K-contiguous; matches the CDNA5
// 16-bit A 16x32 and B 32x16 lane layouts directly).
// Double-buffered LDS, filled by global_load_async_to_lds_b128 (ASYNCcnt):
// the DMA for K-slab t+1 overlaps the 4 (8 if gated) WMMAs of slab t.
// MODE 0: out = acc + bias[n] (+ residual[m,n])            -> f32 (or bf16)
// MODE 1: a = acc_a+bias[n]; g = acc_g+bias[N+n]; a*sigm(g)-> f32   (GLU)
// MODE 2: same but a*silu(g)                               -> bf16  (FFN GLU)
#define BM 128
#define BN 64
#define BK 32

template <int MODE>
__global__ __launch_bounds__(256)
void gemm_bf16_wmma(const unsigned short* __restrict__ A,   // [M,K] bf16
                    const unsigned short* __restrict__ W,   // [N or 2N, K] bf16
                    const float* __restrict__ bias,         // [N or 2N]
                    const float* __restrict__ residual,     // [M,N] or null
                    float* __restrict__ Cf,                 // [M,N] or null
                    unsigned short* __restrict__ Cb,        // [M,N] or null
                    int M, int N, int K) {
  __shared__ __align__(16) unsigned short sA[2][BM * BK];
  __shared__ __align__(16) unsigned short sB[2][BN * BK];
  __shared__ __align__(16) unsigned short sG[2][BN * BK];

  const int tid  = threadIdx.x;
  const int lane = tid & 31;
  const int wave = tid >> 5;
  const int wm   = wave & 3;       // wave row   (4 in M)
  const int wn   = wave >> 2;      // wave col   (2 in N)
  const int m0   = blockIdx.y * BM;
  const int n0   = blockIdx.x * BN;
  const int lrow = lane & 15;
  const int lko  = (lane >> 4) << 4;   // 0 or 16: lanes 16-31 hold K=16..31

  // per-thread 16B copy slot within a 2048-element (64x32) slab
  const int e    = tid * 8;
  const int erow = e >> 5;
  const int ecol = e & 31;

  const f32x8 zv = {};
  f32x8 accA[2][2], accG[2][2];
  #pragma unroll
  for (int i = 0; i < 2; ++i)
    #pragma unroll
    for (int j = 0; j < 2; ++j) { accA[i][j] = zv; accG[i][j] = zv; }

  const int nk = K / BK;

  // async-DMA one K-slab into LDS buffer `buf`
  auto issue = [&](int buf, int k0) {
    async_cp16(&A[(size_t)(m0 + erow) * K + k0 + ecol],       &sA[buf][e]);
    async_cp16(&A[(size_t)(m0 + 64 + erow) * K + k0 + ecol],  &sA[buf][e + 2048]);
    async_cp16(&W[(size_t)(n0 + erow) * K + k0 + ecol],       &sB[buf][e]);
    if (MODE != 0)
      async_cp16(&W[(size_t)(N + n0 + erow) * K + k0 + ecol], &sG[buf][e]);
  };

  issue(0, 0);
  for (int kt = 0; kt < nk; ++kt) {
    const int cur = kt & 1;
    wait_async0();        // slab `cur` landed (issued last iteration / prologue)
    __syncthreads();      // publish DMA writes; all readers of `cur^1` are done
    if (kt + 1 < nk) issue(cur ^ 1, (kt + 1) * BK);  // overlap with WMMAs below

    // fragments: lane<16 -> k 0..15, lane>=16 -> k 16..31 (contiguous k order, same for A and B)
    bf16x16 a0 = *(const bf16x16*)(&sA[cur][(wm * 32 + 0  + lrow) * BK + lko]);
    bf16x16 a1 = *(const bf16x16*)(&sA[cur][(wm * 32 + 16 + lrow) * BK + lko]);
    bf16x16 b0 = *(const bf16x16*)(&sB[cur][(wn * 32 + 0  + lrow) * BK + lko]);
    bf16x16 b1 = *(const bf16x16*)(&sB[cur][(wn * 32 + 16 + lrow) * BK + lko]);

    accA[0][0] = __builtin_amdgcn_wmma_f32_16x16x32_bf16(false, a0, false, b0,
                                       (short)0, accA[0][0], false, false);
    accA[0][1] = __builtin_amdgcn_wmma_f32_16x16x32_bf16(false, a0, false, b1,
                                       (short)0, accA[0][1], false, false);
    accA[1][0] = __builtin_amdgcn_wmma_f32_16x16x32_bf16(false, a1, false, b0,
                                       (short)0, accA[1][0], false, false);
    accA[1][1] = __builtin_amdgcn_wmma_f32_16x16x32_bf16(false, a1, false, b1,
                                       (short)0, accA[1][1], false, false);
    if (MODE != 0) {
      bf16x16 g0 = *(const bf16x16*)(&sG[cur][(wn * 32 + 0  + lrow) * BK + lko]);
      bf16x16 g1 = *(const bf16x16*)(&sG[cur][(wn * 32 + 16 + lrow) * BK + lko]);
      accG[0][0] = __builtin_amdgcn_wmma_f32_16x16x32_bf16(false, a0, false, g0,
                                         (short)0, accG[0][0], false, false);
      accG[0][1] = __builtin_amdgcn_wmma_f32_16x16x32_bf16(false, a0, false, g1,
                                         (short)0, accG[0][1], false, false);
      accG[1][0] = __builtin_amdgcn_wmma_f32_16x16x32_bf16(false, a1, false, g0,
                                         (short)0, accG[1][0], false, false);
      accG[1][1] = __builtin_amdgcn_wmma_f32_16x16x32_bf16(false, a1, false, g1,
                                         (short)0, accG[1][1], false, false);
    }
  }

  // ---- epilogue: C layout lane<16 -> M=r, lane>=16 -> M=8+r; N=lane&15 ----
  #pragma unroll
  for (int i = 0; i < 2; ++i) {
    #pragma unroll
    for (int j = 0; j < 2; ++j) {
      int col     = n0 + wn * 32 + j * 16 + (lane & 15);
      int rowBase = m0 + wm * 32 + i * 16 + ((lane >> 4) << 3);
      #pragma unroll
      for (int r = 0; r < 8; ++r) {
        int row = rowBase + r;
        size_t o = (size_t)row * N + col;
        float va = accA[i][j][r];
        if (MODE == 0) {
          float v = va + (bias ? bias[col] : 0.f) +
                    (residual ? residual[o] : 0.f);
          if (Cf) Cf[o] = v;
          if (Cb) Cb[o] = f2bf(v);
        } else {
          float a = va + bias[col];
          float g = accG[i][j][r] + bias[N + col];
          float sg = 1.f / (1.f + __expf(-g));
          float v = (MODE == 1) ? a * sg : a * g * sg;   // GLU / SiLU-GLU
          if (MODE == 1) Cf[o] = v;
          else           Cb[o] = f2bf(v);
        }
      }
    }
  }
}

// ---------------------------------------------------------------------------
extern "C" void kernel_launch(void* const* d_in, const int* in_sizes, int n_in,
                              void* d_out, int out_size, void* d_ws, size_t ws_size,
                              hipStream_t stream) {
  const int*   x        = (const int*)  d_in[0];
  const float* emb      = (const float*)d_in[2];
  const float* log_dt   = (const float*)d_in[3];
  const float* A_re     = (const float*)d_in[4];
  const float* A_im     = (const float*)d_in[5];
  const float* C_re     = (const float*)d_in[6];
  const float* C_im     = (const float*)d_in[7];
  const float* s4_D     = (const float*)d_in[8];
  const float* s4_out_w = (const float*)d_in[9];
  const float* s4_out_b = (const float*)d_in[10];
  const float* n1w      = (const float*)d_in[11];
  const float* n1b      = (const float*)d_in[12];
  const float* n2w      = (const float*)d_in[13];
  const float* n2b      = (const float*)d_in[14];
  const float* ff_w1    = (const float*)d_in[15];
  const float* ff_b1    = (const float*)d_in[16];
  const float* ff_w2    = (const float*)d_in[17];
  const float* ff_b2    = (const float*)d_in[18];
  const float* pnw      = (const float*)d_in[19];
  const float* pnb      = (const float*)d_in[20];
  const float* out_w    = (const float*)d_in[21];
  const float* out_b    = (const float*)d_in[22];
  float* logits = (float*)d_out;

  // ---- workspace carve-out (~284 MB) ----
  char* ws = (char*)d_ws;
  size_t off = 0;
  auto carve = [&](size_t bytes) -> void* {
    void* p = ws + off;
    off = (off + bytes + 255) & ~(size_t)255;
    return p;
  };
  unsigned short* wb_s4  = (unsigned short*)carve((size_t)NS4c * 2 * DIMc * DIMc * 2);
  unsigned short* wb_f1  = (unsigned short*)carve((size_t)DEPTHc * 2 * FFc * DIMc * 2);
  unsigned short* wb_f2  = (unsigned short*)carve((size_t)DEPTHc * DIMc * FFc * 2);
  unsigned short* wb_out = (unsigned short*)carve((size_t)VOCc * DIMc * 2);
  float*          hbuf   = (float*)carve((size_t)Mrows * DIMc * 4);
  float*          rbuf   = (float*)carve((size_t)Mrows * DIMc * 4);
  unsigned short* ybf    = (unsigned short*)carve((size_t)Mrows * DIMc * 2);
  unsigned short* lnb    = (unsigned short*)carve((size_t)Mrows * DIMc * 2);
  unsigned short* tbf    = (unsigned short*)carve((size_t)Mrows * FFc * 2);

  // ---- one-time (per call) weight downconversion fp32 -> bf16 ----
  cast_f32_bf16_k<<<4096, 256, 0, stream>>>(s4_out_w, wb_s4,
      (size_t)NS4c * 2 * DIMc * DIMc);
  cast_f32_bf16_k<<<4096, 256, 0, stream>>>(ff_w1, wb_f1,
      (size_t)DEPTHc * 2 * FFc * DIMc);
  cast_f32_bf16_k<<<4096, 256, 0, stream>>>(ff_w2, wb_f2,
      (size_t)DEPTHc * DIMc * FFc);
  cast_f32_bf16_k<<<4096, 256, 0, stream>>>(out_w, wb_out,
      (size_t)VOCc * DIMc);

  // ---- embedding gather ----
  gather_k<<<Mrows, 256, 0, stream>>>(x, emb, hbuf);

  for (int i = 0; i < DEPTHc; ++i) {
    // PreNorm(S4 x2) + residual
    layernorm_k<<<Mrows, 256, 0, stream>>>(hbuf, n1w + i * DIMc, n1b + i * DIMc,
                                           rbuf, nullptr);
    for (int j = 0; j < 2; ++j) {
      int k = 2 * i + j;
      s4_scan_k<<<(Bc * DIMc * 32) / 256, 256, 0, stream>>>(
          rbuf, log_dt + (size_t)k * DIMc,
          A_re + (size_t)k * DIMc * NSTc, A_im + (size_t)k * DIMc * NSTc,
          C_re + (size_t)k * DIMc * NSTc, C_im + (size_t)k * DIMc * NSTc,
          s4_D + (size_t)k * DIMc, ybf);
      dim3 g1(DIMc / BN, Mrows / BM);
      gemm_bf16_wmma<1><<<g1, 256, 0, stream>>>(
          ybf, wb_s4 + (size_t)k * 2 * DIMc * DIMc,
          s4_out_b + (size_t)k * 2 * DIMc, nullptr, rbuf, nullptr,
          Mrows, DIMc, DIMc);
    }
    add_k<<<4096, 256, 0, stream>>>(hbuf, rbuf, (size_t)Mrows * DIMc);

    // PreNorm(GLU-FFN) + residual (residual fused in second GEMM)
    layernorm_k<<<Mrows, 256, 0, stream>>>(hbuf, n2w + i * DIMc, n2b + i * DIMc,
                                           nullptr, lnb);
    dim3 g2(FFc / BN, Mrows / BM);
    gemm_bf16_wmma<2><<<g2, 256, 0, stream>>>(
        lnb, wb_f1 + (size_t)i * 2 * FFc * DIMc,
        ff_b1 + (size_t)i * 2 * FFc, nullptr, nullptr, tbf,
        Mrows, FFc, DIMc);
    dim3 g3(DIMc / BN, Mrows / BM);
    gemm_bf16_wmma<0><<<g3, 256, 0, stream>>>(
        tbf, wb_f2 + (size_t)i * DIMc * FFc,
        ff_b2 + (size_t)i * DIMc, hbuf, hbuf, nullptr,
        Mrows, DIMc, FFc);
  }

  // ---- final norm + logits ----
  layernorm_k<<<Mrows, 256, 0, stream>>>(hbuf, pnw, pnb, nullptr, lnb);
  dim3 g4(VOCc / BN, Mrows / BM);   // 500 x 32 workgroups
  gemm_bf16_wmma<0><<<g4, 256, 0, stream>>>(
      lnb, wb_out, out_b, nullptr, logits, nullptr, Mrows, VOCc, DIMc);
}
